// FrameQuantizer_1906965479579
// MI455X (gfx1250) — matrix operand order repaired and checked by reading, not compiled, not run on hardware
//
#include <hip/hip_runtime.h>
#include <hip/hip_bf16.h>
#include <stdint.h>

// ---------------------------------------------------------------------------
// CDNA5 (gfx1250) VQ frame quantizer
//   z   : [8, 8, 256, 1024] fp32
//   emb : [2048, 256] fp32
//   out : z_q [8,8,256,1024] fp32  |  loss (1)  |  indices [8,8,1024] (as fp32)
// ---------------------------------------------------------------------------

typedef __attribute__((ext_vector_type(16))) __bf16 v16bf;
typedef __attribute__((ext_vector_type(8)))  float  v8f;

#define BATCH   8
#define CH      8
#define HH      256           // frame dim (reduction dim d)
#define WW      1024
#define NCODE   2048
#define DIM     256
#define NROW    (BATCH * WW * CH)                // 65536
#define ZQ_ELEMS ((size_t)BATCH * CH * HH * WW)  // 16777216

#define ROWS_PER_BLOCK 128     // 8 waves x 16 rows
#define NTILE (NCODE / 16)     // 128 code tiles
#define B_PITCH  264           // bf16 elems per LDS B row (528B) -> conflict-free b128
#define B_PITCH_B (B_PITCH * 2)

#if __has_builtin(__builtin_amdgcn_global_load_async_to_lds_b128)
#define USE_ASYNC_LDS 1
typedef int v4i __attribute__((vector_size(16)));
typedef __attribute__((address_space(1))) v4i as1_v4i;   // global int4
typedef __attribute__((address_space(3))) v4i as3_v4i;   // LDS int4
#else
#define USE_ASYNC_LDS 0
#endif

// ---------------------------------------------------------------------------
// 1) emb fp32 -> bf16, plus ||e||^2 in fp32
// ---------------------------------------------------------------------------
__global__ __launch_bounds__(256) void pack_emb_kernel(
    const float* __restrict__ emb, __bf16* __restrict__ ebf,
    float* __restrict__ enorm)
{
    const int code = blockIdx.x;
    const int t    = threadIdx.x;            // 0..255 == DIM
    float e = emb[(size_t)code * DIM + t];
    ebf[(size_t)code * DIM + t] = (__bf16)e;
    float s = e * e;
    #pragma unroll
    for (int off = 16; off; off >>= 1) s += __shfl_down(s, off, 32);
    __shared__ float wsum[8];
    if ((t & 31) == 0) wsum[t >> 5] = s;
    __syncthreads();
    if (t == 0) {
        float tot = 0.f;
        #pragma unroll
        for (int i = 0; i < 8; ++i) tot += wsum[i];
        enorm[code] = tot;
    }
}

// ---------------------------------------------------------------------------
// 2) z [b,c,h,w] fp32  ->  A [n][256] bf16, n = (b*W + w)*C + c
//    LDS-tiled 32x32 transpose per (b,c) plane; coalesced read and write.
// ---------------------------------------------------------------------------
__global__ __launch_bounds__(256) void pack_z_kernel(
    const float* __restrict__ z, __bf16* __restrict__ A)
{
    __shared__ float tile[32][33];
    const int tx = threadIdx.x;              // 0..31
    const int ty = threadIdx.y;              // 0..7
    const int bc = blockIdx.z;               // 0..63  (b*8 + c)
    const int h0 = blockIdx.y * 32;
    const int w0 = blockIdx.x * 32;
    const size_t plane = (size_t)bc * HH * WW;

    #pragma unroll
    for (int i = 0; i < 4; ++i) {
        int h = h0 + ty + i * 8;
        tile[ty + i * 8][tx] = z[plane + (size_t)h * WW + (w0 + tx)];
    }
    __syncthreads();

    const int b = bc >> 3, c = bc & 7;
    #pragma unroll
    for (int i = 0; i < 4; ++i) {
        int w    = w0 + ty + i * 8;
        int nrow = (b * WW + w) * CH + c;
        A[(size_t)nrow * DIM + (h0 + tx)] = (__bf16)tile[tx][ty + i * 8];
    }
}

// ---------------------------------------------------------------------------
// Stage one B tile (16 codes x 256 dims bf16 = 8KB, pitch-padded in LDS).
// Async path: per-lane GLOBAL_LOAD_ASYNC_TO_LDS_B128 (tracked on ASYNCcnt).
// ---------------------------------------------------------------------------
__device__ __forceinline__ void stage_b_tile(const __bf16* __restrict__ ebf,
                                             __bf16* buf, int tile, int tid)
{
    const char* src = (const char*)ebf + (size_t)tile * 16 * 512;  // 16 rows x 512B
    char* dst = (char*)buf;
#if USE_ASYNC_LDS
    #pragma unroll
    for (int i = 0; i < 2; ++i) {
        int c = tid * 2 + i;                 // chunk 0..511 (16B each)
        int r = c >> 5, j = c & 31;
        __builtin_amdgcn_global_load_async_to_lds_b128(
            (as1_v4i*)(src + r * 512 + j * 16),
            (as3_v4i*)(dst + r * B_PITCH_B + j * 16), 0, 0);
    }
#else
    for (int i = tid; i < 512; i += 256) {
        int r = i >> 5, j = i & 31;
        uint4 v = *(const uint4*)(src + r * 512 + j * 16);
        *(uint4*)(dst + r * B_PITCH_B + j * 16) = v;
    }
#endif
}

__device__ __forceinline__ void async_wait_2()
{
#if USE_ASYNC_LDS
# if __has_builtin(__builtin_amdgcn_s_wait_asynccnt)
    __builtin_amdgcn_s_wait_asynccnt(2);
# else
    asm volatile("s_wait_asynccnt 0x2" ::: "memory");
# endif
#endif
}

__device__ __forceinline__ void async_wait_0()
{
#if USE_ASYNC_LDS
# if __has_builtin(__builtin_amdgcn_s_wait_asynccnt)
    __builtin_amdgcn_s_wait_asynccnt(0);
# else
    asm volatile("s_wait_asynccnt 0x0" ::: "memory");
# endif
#endif
}

// ---------------------------------------------------------------------------
// 3) Fused GEMM + argmin.
//    Block = 256 thr: 8 waves x 16 private rows (A frags register-resident,
//    loaded straight from global). All waves sweep all 128 code tiles; the
//    B tile is double-buffered in LDS via async global->LDS copies. Per tile:
//    8x v_wmma_f32_16x16x32_bf16 (K=256), epilogue score = ||e||^2 - 2*dot,
//    packed (orderable-float || code) u64, wave-local shfl_xor min reduction.
// ---------------------------------------------------------------------------
__global__ __launch_bounds__(256) void vq_argmin_kernel(
    const __bf16* __restrict__ A,        // [NROW][DIM]
    const __bf16* __restrict__ ebf,      // [NCODE][DIM]
    const float*  __restrict__ enorm,    // [NCODE]
    unsigned*     __restrict__ best_idx) // [NROW]
{
    __shared__ __bf16 sB[2][16 * B_PITCH];   // 2 x 8.25KB double buffer

    const int tid     = threadIdx.x;
    const int lane    = tid & 31;
    const int wave    = tid >> 5;
    const int m       = lane & 15;            // row-in-strip / C column index
    const int half    = lane >> 4;
    const int rowBase = blockIdx.x * ROWS_PER_BLOCK + wave * 16;

    union Frag { v16bf v; uint4 q[2]; };

    // A fragments (ISA 16-bit A 16x32 layout): lane<16 -> K {0..7,16..23},
    // lane>=16 -> K {8..15,24..31}; two 16B chunks per 32-wide K block.
    Frag a[8];
    {
        const __bf16* arow = A + (size_t)(rowBase + m) * DIM;
        #pragma unroll
        for (int kb = 0; kb < 8; ++kb) {
            a[kb].q[0] = *(const uint4*)(arow + kb * 32 + half * 8);
            a[kb].q[1] = *(const uint4*)(arow + kb * 32 + half * 8 + 16);
        }
    }

    unsigned long long best[8];
    #pragma unroll
    for (int r = 0; r < 8; ++r) best[r] = 0xFFFFFFFFFFFFFFFFull;

    stage_b_tile(ebf, sB[0], 0, tid);

    for (int t = 0; t < NTILE; ++t) {
        const int cur = t & 1;
        if (t + 1 < NTILE) {
            stage_b_tile(ebf, sB[cur ^ 1], t + 1, tid);
            async_wait_2();                  // tile t's chunks done, t+1 in flight
        } else {
            async_wait_0();
        }
        __syncthreads();                     // all waves' chunks visible

        const int code = t * 16 + m;         // this lane's output column
        const float en = enorm[code];
        v8f acc = {};
        #pragma unroll
        for (int kb = 0; kb < 8; ++kb) {
            // B (32x16): lane holds column n=m, K {half*8..+7, 16+half*8..+7}.
            Frag bf;
            const __bf16* q0 = &sB[cur][m * B_PITCH + kb * 32 + half * 8];
            bf.q[0] = *(const uint4*)q0;
            bf.q[1] = *(const uint4*)(q0 + 16);
            acc = __builtin_amdgcn_wmma_f32_16x16x32_bf16(
                      false, a[kb].v, false, bf.v, (short)0, acc, false, false);
        }
        #pragma unroll
        for (int r = 0; r < 8; ++r) {
            float score = fmaf(-2.0f, acc[r], en);    // ||x||^2 const per row
            unsigned u = __float_as_uint(score);
            u = (u & 0x80000000u) ? ~u : (u | 0x80000000u);   // orderable key
            unsigned long long packed =
                ((unsigned long long)u << 32) | (unsigned)code;
            if (packed < best[r]) best[r] = packed;   // tie -> lowest code
        }
        __syncthreads();                     // sB[cur] free before next issue
    }

    // Wave-local argmin: reduce over the 16 lanes of each half (xor net keeps
    // halves disjoint). Lane m==0 of each half holds the minimum.
    #pragma unroll
    for (int r = 0; r < 8; ++r) {
        unsigned long long b = best[r];
        #pragma unroll
        for (int off = 8; off; off >>= 1) {
            unsigned long long o = __shfl_xor(b, off, 32);
            if (o < b) b = o;
        }
        if (m == 0)
            best_idx[rowBase + half * 8 + r] = (unsigned)(b & 0xFFFFFFFFu);
    }
}

// ---------------------------------------------------------------------------
// 4) Gather z_q, write indices, reduce exact fp32 loss sum.
// ---------------------------------------------------------------------------
__global__ __launch_bounds__(256) void scatter_kernel(
    const float* __restrict__ z, const float* __restrict__ emb,
    const unsigned* __restrict__ best_idx,
    float* __restrict__ out, float* __restrict__ loss_accum)
{
    const int tx = threadIdx.x;
    const int w  = blockIdx.x * 256 + tx;
    const int bc = blockIdx.y;               // b*8 + c
    const int b  = bc >> 3, c = bc & 7;

    const unsigned idx = best_idx[(size_t)(b * WW + w) * CH + c];
    out[ZQ_ELEMS + 1 + (size_t)bc * WW + w] = (float)idx;   // indices [b,c,w]

    const float4* erow = (const float4*)(emb + (size_t)idx * DIM);
    const size_t zbase = (size_t)bc * HH * WW + w;
    float lsum = 0.f;
    #pragma unroll 4
    for (int h4 = 0; h4 < HH / 4; ++h4) {
        float4 e4 = erow[h4];                 // L2-hot gather (emb = 2MB)
        float ev[4] = {e4.x, e4.y, e4.z, e4.w};
        #pragma unroll
        for (int j = 0; j < 4; ++j) {
            size_t off = zbase + (size_t)(h4 * 4 + j) * WW;
            float zv = z[off];
            out[off]  = ev[j];                // z_q [b,c,h,w], coalesced
            float d = ev[j] - zv;
            lsum = fmaf(d, d, lsum);
        }
    }
    #pragma unroll
    for (int off = 16; off; off >>= 1) lsum += __shfl_down(lsum, off, 32);
    __shared__ float wsum[8];
    if ((tx & 31) == 0) wsum[tx >> 5] = lsum;
    __syncthreads();
    if (tx == 0) {
        float s = 0.f;
        #pragma unroll
        for (int i = 0; i < 8; ++i) s += wsum[i];
        atomicAdd(loss_accum, s);
    }
}

__global__ void init_kernel(float* loss_accum) { *loss_accum = 0.f; }

__global__ void finalize_kernel(const float* loss_accum, float* out_loss) {
    // loss = mean + 0.25*mean of the identical residual = 1.25 * mean
    *out_loss = 1.25f * (*loss_accum) / (float)ZQ_ELEMS;
}

// ---------------------------------------------------------------------------
extern "C" void kernel_launch(void* const* d_in, const int* in_sizes, int n_in,
                              void* d_out, int out_size, void* d_ws, size_t ws_size,
                              hipStream_t stream)
{
    const float* z   = (const float*)d_in[0];
    const float* emb = (const float*)d_in[1];
    float* out = (float*)d_out;

    char* ws = (char*)d_ws;
    __bf16*   A        = (__bf16*)ws;   ws += (size_t)NROW  * DIM * 2;  // 32 MB
    __bf16*   ebf      = (__bf16*)ws;   ws += (size_t)NCODE * DIM * 2;  //  1 MB
    float*    enorm    = (float*)ws;    ws += (size_t)NCODE * 4;
    unsigned* best_idx = (unsigned*)ws; ws += (size_t)NROW * 4;
    float*    loss_acc = (float*)ws;

    init_kernel<<<1, 1, 0, stream>>>(loss_acc);
    pack_emb_kernel<<<NCODE, 256, 0, stream>>>(emb, ebf, enorm);
    pack_z_kernel<<<dim3(WW / 32, HH / 32, BATCH * CH), dim3(32, 8), 0, stream>>>(z, A);
    vq_argmin_kernel<<<NROW / ROWS_PER_BLOCK, 256, 0, stream>>>(A, ebf, enorm, best_idx);
    scatter_kernel<<<dim3(WW / 256, BATCH * CH), 256, 0, stream>>>(
        z, emb, best_idx, out, loss_acc);
    finalize_kernel<<<1, 1, 0, stream>>>(loss_acc, out + ZQ_ELEMS);
}
